// GINEncoder_55920474194401
// MI455X (gfx1250) — matrix operand order repaired and checked by reading, not compiled, not run on hardware
//
#include <hip/hip_runtime.h>
#include <hip/hip_bf16.h>
#include <stdint.h>

// ---------------------------------------------------------------------------
// GIN encoder: 3 x (segment_sum aggregation + MLP) on MI455X (gfx1250).
// GEMMs: v_wmma_f32_16x16x32_bf16, wave tile 16x64 (4 accumulators),
// double-buffered LDS A staging (GLOBAL_LOAD_ASYNC_TO_LDS_B128 for bf16 A),
// fully unrolled K loop. Aggregation: non-returning global_atomic_add_f32.
// ---------------------------------------------------------------------------

#define GIN_N   100000
#define GIN_E   640000

#if __has_builtin(__builtin_amdgcn_global_load_async_to_lds_b128) && \
    __has_builtin(__builtin_amdgcn_s_wait_asynccnt)
#define GIN_ASYNC_LDS 1
#else
#define GIN_ASYNC_LDS 0
#endif

typedef __attribute__((ext_vector_type(4)))  unsigned short us4;
typedef __attribute__((ext_vector_type(8)))  unsigned short us8;
typedef __attribute__((ext_vector_type(16))) unsigned short us16;
typedef __attribute__((ext_vector_type(16))) __bf16        v16bf;
typedef __attribute__((ext_vector_type(8)))  float         v8f;

// builtin_amdgcn_global_load_async_to_lds_b128 expects pointers to this type
typedef int gin_v4i __attribute__((vector_size(16)));

union FragU { us16 u; v16bf b; };

__device__ __forceinline__ unsigned short f2bf(float f) {
    union { float f; unsigned u; } v; v.f = f;
    unsigned u = v.u;
    // round-to-nearest-even truncation to bf16
    return (unsigned short)((u + 0x7FFFu + ((u >> 16) & 1u)) >> 16);
}

__device__ __forceinline__ void wait_async_lds() {
#if GIN_ASYNC_LDS
    __builtin_amdgcn_s_wait_asynccnt(0);
#endif
}

// ---------------------------------------------------------------------------
// Weight convert: W[K][Nout] fp32 row-major -> Wt[Nout][K] bf16 (transposed)
// ---------------------------------------------------------------------------
__global__ void wcvt_kernel(const float* __restrict__ w,
                            unsigned short* __restrict__ wt,
                            int K, int noutShift) {
    int tid = blockIdx.x * blockDim.x + threadIdx.x;
    if (tid >= (K << noutShift)) return;
    int n = tid & ((1 << noutShift) - 1);
    int k = tid >> noutShift;
    wt[(size_t)n * K + k] = f2bf(w[tid]);
}

// ---------------------------------------------------------------------------
// agg = x  (float4 copy; seeds the GIN self-term so atomics fuse x + agg)
// ---------------------------------------------------------------------------
__global__ void copy4_kernel(float* __restrict__ dst,
                             const float* __restrict__ src, int n4) {
    int i = blockIdx.x * blockDim.x + threadIdx.x;
    if (i < n4) ((float4*)dst)[i] = ((const float4*)src)[i];
}

// ---------------------------------------------------------------------------
// agg[dst] += x[src], 4 floats per thread. shift: log2(d/4) (5 -> d=128,
// 6 -> d=256). x (51-102 MB) is L2-resident; relaxed agent-scope fetch_add
// lowers to non-returning global_atomic_add_f32.
// ---------------------------------------------------------------------------
__global__ void agg_edges_kernel(float* __restrict__ agg,
                                 const float* __restrict__ x,
                                 const long long* __restrict__ ei,
                                 int E, int shift) {
    long long tid = (long long)blockIdx.x * blockDim.x + threadIdx.x;
    long long total = (long long)E << shift;
    if (tid >= total) return;
    int e = (int)(tid >> shift);
    int q = (int)(tid & ((1 << shift) - 1));
    int d = 4 << shift;
    long long s = ei[e];
    long long t = ei[E + e];
    const float4 v = *(const float4*)(x + (size_t)s * d + ((size_t)q << 2));
    float* o = agg + (size_t)t * d + ((size_t)q << 2);
    __hip_atomic_fetch_add(o + 0, v.x, __ATOMIC_RELAXED, __HIP_MEMORY_SCOPE_AGENT);
    __hip_atomic_fetch_add(o + 1, v.y, __ATOMIC_RELAXED, __HIP_MEMORY_SCOPE_AGENT);
    __hip_atomic_fetch_add(o + 2, v.z, __ATOMIC_RELAXED, __HIP_MEMORY_SCOPE_AGENT);
    __hip_atomic_fetch_add(o + 3, v.w, __ATOMIC_RELAXED, __HIP_MEMORY_SCOPE_AGENT);
}

// ---------------------------------------------------------------------------
// Tiled GEMM + bias + ReLU on v_wmma_f32_16x16x32_bf16.
//   Out[M,NOUT] = relu(A[M,K] @ W[K,NOUT] + bias),  M = 100000
//   A_F32:    A is fp32 (bf16-converted while staging to LDS), else bf16
//             (staged via GLOBAL_LOAD_ASYNC_TO_LDS_B128 when available).
//   OUT_BF16: emit bf16 (feeds next WMMA), else fp32.
// Block: 256 threads = 8 waves as 4 row-waves x 2 col-waves.
// Block tile 64 x 128; wave tile 16 x 64 -> 4 accumulators per wave, one
// A fragment (2x ds_read_b128) feeds 4 WMMAs. A tile is double-buffered in
// LDS (stride 40 shorts -> conflict-free b128 reads), 1 barrier / K-step.
// All 4 B fragments are loaded before the WMMAs (clause-able), staging for
// the next K-step is issued before the WMMAs execute. K loop fully unrolled.
// Bias folded into accumulator init. A and B fragments use contiguous k order (k = 16*hi + s).
// ---------------------------------------------------------------------------
template <int K, int NOUT, bool A_F32, bool OUT_BF16>
__global__ __launch_bounds__(256) void gemm_bias_relu_kernel(
        const void* __restrict__ Av,
        const unsigned short* __restrict__ Wt,
        const float* __restrict__ bias,
        void* __restrict__ Outv) {
    constexpr int M = GIN_N;
    constexpr int LDST = 40;            // padded LDS row stride (shorts)
    constexpr int BUFS = 64 * LDST;     // one A buffer (shorts)
    __shared__ alignas(16) unsigned short As[2 * BUFS];

    const int t    = threadIdx.x;
    const int lane = t & 31;
    const int m    = lane & 15;
    const int hi   = lane >> 4;         // 0 or 1 (lane half)
    const int wr   = (t >> 5) & 3;      // row-wave 0..3
    const int wc   = t >> 7;            // col-wave 0..1

    const int rowBase = blockIdx.x * 64 + wr * 16;
    const int colBase = blockIdx.y * 128 + wc * 64;

    // staging coords: 64 rows x 32 k, 8 shorts (16B) per thread
    const int sr = t >> 2;              // 0..63
    const int sc = (t & 3) << 3;        // 0,8,16,24
    const int gr = blockIdx.x * 64 + sr;
    const bool rowOk = gr < M;

    // B row pointers (4 col fragments per wave)
    const unsigned short* wrow0 = Wt + (size_t)(colBase +  0 + m) * K + (hi << 4);
    const unsigned short* wrow1 = Wt + (size_t)(colBase + 16 + m) * K + (hi << 4);
    const unsigned short* wrow2 = Wt + (size_t)(colBase + 32 + m) * K + (hi << 4);
    const unsigned short* wrow3 = Wt + (size_t)(colBase + 48 + m) * K + (hi << 4);

    // bias folded into accumulator init (per-lane column constant)
    const float bv0 = bias[colBase +  0 + m];
    const float bv1 = bias[colBase + 16 + m];
    const float bv2 = bias[colBase + 32 + m];
    const float bv3 = bias[colBase + 48 + m];
    v8f acc0 = {bv0, bv0, bv0, bv0, bv0, bv0, bv0, bv0};
    v8f acc1 = {bv1, bv1, bv1, bv1, bv1, bv1, bv1, bv1};
    v8f acc2 = {bv2, bv2, bv2, bv2, bv2, bv2, bv2, bv2};
    v8f acc3 = {bv3, bv3, bv3, bv3, bv3, bv3, bv3, bv3};

    const unsigned short* alds = As + (size_t)(wr * 16 + m) * LDST + (hi << 4);

    auto stage = [&](int k0, int p) {
        unsigned short* dst = As + p * BUFS + sr * LDST + sc;
        if constexpr (A_F32) {
            const float* A = (const float*)Av;
            float4 v0 = {0.f, 0.f, 0.f, 0.f};
            float4 v1 = {0.f, 0.f, 0.f, 0.f};
            if (rowOk) {
                const float* src = A + (size_t)gr * K + k0 + sc;
                v0 = *(const float4*)(src);
                v1 = *(const float4*)(src + 4);
            }
            us8 h;
            h[0] = f2bf(v0.x); h[1] = f2bf(v0.y);
            h[2] = f2bf(v0.z); h[3] = f2bf(v0.w);
            h[4] = f2bf(v1.x); h[5] = f2bf(v1.y);
            h[6] = f2bf(v1.z); h[7] = f2bf(v1.w);
            *(us8*)dst = h;
        } else {
            const unsigned short* A = (const unsigned short*)Av;
            if (rowOk) {
                const unsigned short* src = A + (size_t)gr * K + k0 + sc;
#if GIN_ASYNC_LDS
                __builtin_amdgcn_global_load_async_to_lds_b128(
                    (__attribute__((address_space(1))) gin_v4i*)src,
                    (__attribute__((address_space(3))) gin_v4i*)dst,
                    0, 0);
#else
                *(us8*)dst = *(const us8*)src;
#endif
            } else {
                us8 z = {};
                *(us8*)dst = z;
            }
        }
    };

    stage(0, 0);
    if constexpr (!A_F32) wait_async_lds();
    __syncthreads();

    int p = 0;
#pragma unroll
    for (int k0 = 0; k0 < K; k0 += 32) {
        const bool more = (k0 + 32) < K;

        // A fragment (LDS) and all 4 B fragments (global, L2-resident) first
        const unsigned short* ap = alds + p * BUFS;
        us8 alo = *(const us8*)(ap);
        us8 ahi = *(const us8*)(ap + 8);
        FragU fa;
        fa.u = __builtin_shufflevector(alo, ahi,
                 0, 1, 2, 3, 4, 5, 6, 7, 8, 9, 10, 11, 12, 13, 14, 15);
        FragU fb0; fb0.u = *(const us16*)(wrow0 + k0);
        FragU fb1; fb1.u = *(const us16*)(wrow1 + k0);
        FragU fb2; fb2.u = *(const us16*)(wrow2 + k0);
        FragU fb3; fb3.u = *(const us16*)(wrow3 + k0);

        // issue next tile's staging before the WMMAs (overlap)
        if (more) stage(k0 + 32, p ^ 1);

        acc0 = __builtin_amdgcn_wmma_f32_16x16x32_bf16(
                   false, fa.b, false, fb0.b, (short)0, acc0, false, false);
        acc1 = __builtin_amdgcn_wmma_f32_16x16x32_bf16(
                   false, fa.b, false, fb1.b, (short)0, acc1, false, false);
        acc2 = __builtin_amdgcn_wmma_f32_16x16x32_bf16(
                   false, fa.b, false, fb2.b, (short)0, acc2, false, false);
        acc3 = __builtin_amdgcn_wmma_f32_16x16x32_bf16(
                   false, fa.b, false, fb3.b, (short)0, acc3, false, false);

        if (more) {
            if constexpr (!A_F32) wait_async_lds();
            __syncthreads();
        }
        p ^= 1;
    }

    // epilogue: ReLU + store (bias already accumulated)
    const int orow0 = rowBase + (hi << 3);
    auto emit = [&](const v8f& a, int col) {
#pragma unroll
        for (int r = 0; r < 8; ++r) {
            const int row = orow0 + r;
            if (row < M) {
                float v = a[r];
                v = v > 0.f ? v : 0.f;
                if constexpr (OUT_BF16)
                    ((unsigned short*)Outv)[(size_t)row * NOUT + col] = f2bf(v);
                else
                    ((float*)Outv)[(size_t)row * NOUT + col] = v;
            }
        }
    };
    emit(acc0, colBase +  0 + m);
    emit(acc1, colBase + 16 + m);
    emit(acc2, colBase + 32 + m);
    emit(acc3, colBase + 48 + m);
}

// ---------------------------------------------------------------------------
// Launcher
// ---------------------------------------------------------------------------
extern "C" void kernel_launch(void* const* d_in, const int* in_sizes, int n_in,
                              void* d_out, int out_size, void* d_ws,
                              size_t ws_size, hipStream_t stream) {
    (void)in_sizes; (void)n_in; (void)out_size; (void)ws_size;

    const float*     x0 = (const float*)d_in[0];
    const long long* ei = (const long long*)d_in[1];
    const float* w0a = (const float*)d_in[2];
    const float* b0a = (const float*)d_in[3];
    const float* w0b = (const float*)d_in[4];
    const float* b0b = (const float*)d_in[5];
    const float* w1a = (const float*)d_in[6];
    const float* b1a = (const float*)d_in[7];
    const float* w1b = (const float*)d_in[8];
    const float* b1b = (const float*)d_in[9];
    const float* w2a = (const float*)d_in[10];
    const float* b2a = (const float*)d_in[11];
    const float* w2b = (const float*)d_in[12];
    const float* b2b = (const float*)d_in[13];

    // workspace layout
    char* ws = (char*)d_ws;
    float*          xbuf = (float*)ws;                         // N*256 f32
    float*          agg  = (float*)(ws + 102400000);           // N*256 f32
    unsigned short* hbuf = (unsigned short*)(ws + 204800000);  // N*256 bf16
    unsigned short* wt   = (unsigned short*)(ws + 256000000);  // 278528 bf16

    unsigned short* w0a_t = wt + 0;       // 128x256 -> [256][128]
    unsigned short* w0b_t = wt + 32768;   // 256x256
    unsigned short* w1a_t = wt + 98304;   // 256x256
    unsigned short* w1b_t = wt + 163840;  // 256x256
    unsigned short* w2a_t = wt + 229376;  // 256x128 -> [128][256]
    unsigned short* w2b_t = wt + 262144;  // 128x128

    const int B = 256;
    const int GX = (GIN_N + 63) / 64;     // 1563 row tiles

    // --- weight conversion (transposed bf16), once per launch -------------
    wcvt_kernel<<<128, B, 0, stream>>>(w0a, w0a_t, 128, 8);
    wcvt_kernel<<<256, B, 0, stream>>>(w0b, w0b_t, 256, 8);
    wcvt_kernel<<<256, B, 0, stream>>>(w1a, w1a_t, 256, 8);
    wcvt_kernel<<<256, B, 0, stream>>>(w1b, w1b_t, 256, 8);
    wcvt_kernel<<<128, B, 0, stream>>>(w2a, w2a_t, 256, 7);
    wcvt_kernel<<< 64, B, 0, stream>>>(w2b, w2b_t, 128, 7);

    // --- layer 0: d=128 -> 256 -> 256 -------------------------------------
    copy4_kernel<<<12500, B, 0, stream>>>(agg, x0, GIN_N * 128 / 4);
    agg_edges_kernel<<<80000, B, 0, stream>>>(agg, x0, ei, GIN_E, 5);
    gemm_bias_relu_kernel<128, 256, true, true>
        <<<dim3(GX, 2), B, 0, stream>>>(agg, w0a_t, b0a, hbuf);
    gemm_bias_relu_kernel<256, 256, false, false>
        <<<dim3(GX, 2), B, 0, stream>>>(hbuf, w0b_t, b0b, xbuf);

    // --- layer 1: d=256 -> 256 -> 256 -------------------------------------
    copy4_kernel<<<25000, B, 0, stream>>>(agg, xbuf, GIN_N * 256 / 4);
    agg_edges_kernel<<<160000, B, 0, stream>>>(agg, xbuf, ei, GIN_E, 6);
    gemm_bias_relu_kernel<256, 256, true, true>
        <<<dim3(GX, 2), B, 0, stream>>>(agg, w1a_t, b1a, hbuf);
    gemm_bias_relu_kernel<256, 256, false, false>
        <<<dim3(GX, 2), B, 0, stream>>>(hbuf, w1b_t, b1b, xbuf);

    // --- layer 2: d=256 -> 128 -> 128 (final fp32 to d_out) ---------------
    copy4_kernel<<<25000, B, 0, stream>>>(agg, xbuf, GIN_N * 256 / 4);
    agg_edges_kernel<<<160000, B, 0, stream>>>(agg, xbuf, ei, GIN_E, 6);
    gemm_bias_relu_kernel<256, 128, true, true>
        <<<dim3(GX, 1), B, 0, stream>>>(agg, w2a_t, b2a, hbuf);
    gemm_bias_relu_kernel<128, 128, false, false>
        <<<dim3(GX, 1), B, 0, stream>>>(hbuf, w2b_t, b2b, (float*)d_out);
}
